// CausalSelfAttention_47433618817181
// MI455X (gfx1250) — matrix operand order repaired and checked by reading, not compiled, not run on hardware
//
#include <hip/hip_runtime.h>
#include <hip/hip_bf16.h>
#include <stdint.h>

typedef __attribute__((ext_vector_type(16))) _Float16 v16h;
typedef __attribute__((ext_vector_type(8)))  _Float16 v8h;
typedef __attribute__((ext_vector_type(8)))  float    v8f;

union AFrag { v16h v; v8h h[2]; };

constexpr int cB   = 4;
constexpr int cS   = 2048;
constexpr int cH   = 14;
constexpr int cKV  = 2;
constexpr int cD   = 64;
constexpr int cHID = cH * cD;       // 896
constexpr int cT   = cB * cS;       // 8192
constexpr int cNKV = cKV * cD;      // 128
constexpr float cROPE = 0.431734776f; // ln(1e6)/32

static __device__ __forceinline__ v8f wmma_f16(v16h a, v16h b, v8f c) {
  return __builtin_amdgcn_wmma_f32_16x16x32_f16(false, a, false, b, (short)0, c,
                                                false, false);
}

// ---------------------------------------------------------------- converts
__global__ void cvt_f32_f16(const float* __restrict__ in,
                            _Float16* __restrict__ out, int n) {
  int i = blockIdx.x * blockDim.x + threadIdx.x;
  int stride = gridDim.x * blockDim.x;
  for (; i < n; i += stride) out[i] = (_Float16)in[i];
}

// out[n*K + k] = (f16) in[k*N + n]   (weight [K][N] -> W^T [N][K])
__global__ void transpose_cvt(const float* __restrict__ in,
                              _Float16* __restrict__ out, int K, int N) {
  int i = blockIdx.x * blockDim.x + threadIdx.x;
  int total = K * N;
  int stride = gridDim.x * blockDim.x;
  for (; i < total; i += stride) {
    int n = i / K;
    int k = i - n * K;
    out[i] = (_Float16)in[k * N + n];
  }
}

// ---------------------------------------------------------------- WMMA GEMM
// C[M,N] = A[M,K] (f16, row-major) * Wt[N,K]^T (f16)  + epilogue
// 256 threads = 8 waves; block tile 128x128; wave tile 32(M) x 64(N).
enum { MODE_Q = 0, MODE_K = 1, MODE_V = 2, MODE_O = 3 };

template <int MODE>
__global__ __launch_bounds__(256) void gemm_wmma(
    const _Float16* __restrict__ A, const _Float16* __restrict__ Wt,
    const float* __restrict__ bias, void* __restrict__ outp, int M, int N,
    int K, const int* __restrict__ pos) {
  const int tid = threadIdx.x;
  const int lane = tid & 31;
  const int wid = tid >> 5;
  const int l15 = lane & 15;
  const int hi = lane >> 4;
  const int rowBase = blockIdx.y * 128 + (wid >> 1) * 32;
  const int colBase = blockIdx.x * 128 + (wid & 1) * 64;

  v8f acc[2][4];
#pragma unroll
  for (int mt = 0; mt < 2; ++mt)
#pragma unroll
    for (int nt = 0; nt < 4; ++nt)
#pragma unroll
      for (int e = 0; e < 8; ++e) acc[mt][nt][e] = 0.0f;

  for (int k0 = 0; k0 < K; k0 += 32) {
    AFrag a[2];
#pragma unroll
    for (int mt = 0; mt < 2; ++mt) {
      const _Float16* ap =
          A + (size_t)(rowBase + mt * 16 + l15) * K + k0 + hi * 16;
      a[mt].h[0] = *(const v8h*)ap;
      a[mt].h[1] = *(const v8h*)(ap + 8);
    }
    v16h b[4];
#pragma unroll
    for (int nt = 0; nt < 4; ++nt)
      b[nt] = *(const v16h*)(Wt + (size_t)(colBase + nt * 16 + l15) * K + k0 +
                             hi * 16);
#pragma unroll
    for (int mt = 0; mt < 2; ++mt)
#pragma unroll
      for (int nt = 0; nt < 4; ++nt)
        acc[mt][nt] = wmma_f16(a[mt].v, b[nt], acc[mt][nt]);
  }

  if constexpr (MODE == MODE_O) {
    float* out = (float*)outp;
#pragma unroll
    for (int mt = 0; mt < 2; ++mt) {
      int r0 = rowBase + mt * 16 + hi * 8;
#pragma unroll
      for (int nt = 0; nt < 4; ++nt) {
        int n = colBase + nt * 16 + l15;
#pragma unroll
        for (int v = 0; v < 8; ++v)
          out[(size_t)(r0 + v) * N + n] = acc[mt][nt][v];
      }
    }
    return;
  }

  // bias add
  float bn[4];
#pragma unroll
  for (int nt = 0; nt < 4; ++nt) bn[nt] = bias[colBase + nt * 16 + l15];
#pragma unroll
  for (int mt = 0; mt < 2; ++mt)
#pragma unroll
    for (int nt = 0; nt < 4; ++nt)
#pragma unroll
      for (int v = 0; v < 8; ++v) acc[mt][nt][v] += bn[nt];

  _Float16* out16 = (_Float16*)outp;
#pragma unroll
  for (int mt = 0; mt < 2; ++mt) {
    int r0 = rowBase + mt * 16 + hi * 8;
#pragma unroll
    for (int nt = 0; nt < 4; ++nt) {
      int n = colBase + nt * 16 + l15;
      int head = n >> 6;
      int d = n & 63;
      if constexpr (MODE == MODE_V) {
#pragma unroll
        for (int v = 0; v < 8; ++v) {
          int t = r0 + v;
          int bb = t >> 11;           // t / S
          int srow = t & (cS - 1);    // t % S
          // V transposed: [B][KV][D][S]
          out16[(((size_t)bb * cKV + head) * cD + d) * cS + srow] =
              (_Float16)acc[mt][nt][v];
        }
      } else {
        float invf = __expf(-(float)(d & 31) * cROPE);
#pragma unroll
        for (int v = 0; v < 8; ++v) {
          int t = r0 + v;
          float ang = (float)pos[t] * invf;
          float sn, cs;
          __sincosf(ang, &sn, &cs);
          float partner = acc[mt][nt ^ 2][v];  // column n +/- 32, same row
          float val =
              acc[mt][nt][v] * cs + ((d < 32) ? -partner : partner) * sn;
          int bb = t >> 11;
          int srow = t & (cS - 1);
          if constexpr (MODE == MODE_Q) {
            // fold softmax scale 1/sqrt(64) into Q; layout [B][H][S][D]
            out16[(((size_t)bb * cH + head) * cS + srow) * cD + d] =
                (_Float16)(val * 0.125f);
          } else {
            // K layout [B][KV][S][D]
            out16[(((size_t)bb * cKV + head) * cS + srow) * cD + d] =
                (_Float16)val;
          }
        }
      }
    }
  }
}

// ---------------------------------------------------------------- attention
// Flash attention, causal, transposed-score formulation:
//   S^T = K Q^T   (keys = WMMA rows, queries = WMMA cols)
//   O^T = V^T P   (d    = WMMA rows, queries = WMMA cols)
// A query column lives in only 2 lanes -> softmax reductions are in-lane
// + one shfl_xor(16); running stats are per-lane scalars.
// 128 threads = 4 independent waves, each wave owns 32 query rows.
// Q pre-scaled by 1/sqrt(D). V stored transposed [B][KV][D][S].
__global__ __launch_bounds__(128) void attn_wmma(
    const _Float16* __restrict__ Q, const _Float16* __restrict__ Kc,
    const _Float16* __restrict__ Vt, _Float16* __restrict__ O) {
  __shared__ __align__(32) _Float16 Plds[4][32][32];  // [wave][query][key]

  const int tid = threadIdx.x;
  const int lane = tid & 31;
  const int wid = tid >> 5;
  const int l15 = lane & 15;
  const int hi = lane >> 4;

  const int bh = blockIdx.y;
  const int b = bh / cH;
  const int h = bh % cH;
  const int kvh = h / (cH / cKV);
  const int qbase = blockIdx.x * 128 + wid * 32;

  const _Float16* Qp = Q + (((size_t)b * cH + h) * cS) * cD;
  const _Float16* Kp = Kc + (((size_t)b * cKV + kvh) * cS) * cD;
  const _Float16* Vp = Vt + (((size_t)b * cKV + kvh) * cD) * cS;

  // Q as loop-invariant B-fragments: lane = query col, slots = d
  v16h bq[2][2];
#pragma unroll
  for (int n2 = 0; n2 < 2; ++n2) {
    const _Float16* qp = Qp + (size_t)(qbase + n2 * 16 + l15) * cD;
#pragma unroll
    for (int kd = 0; kd < 2; ++kd)
      bq[n2][kd] = *(const v16h*)(qp + kd * 32 + hi * 16);
  }

  v8f acco[4][2];  // O^T: [d-tile][query-tile]
#pragma unroll
  for (int mt = 0; mt < 4; ++mt)
#pragma unroll
    for (int n2 = 0; n2 < 2; ++n2)
#pragma unroll
      for (int e = 0; e < 8; ++e) acco[mt][n2][e] = 0.0f;
  float mrun[2] = {-1e30f, -1e30f};
  float lrun[2] = {0.0f, 0.0f};

  const int ktLast = qbase / 32;  // wave-uniform causal bound
  for (int kt = 0; kt <= ktLast; ++kt) {
    const int k0 = kt * 32;
    // prefetch next key tile (speculative; dropped if unmapped)
    __builtin_prefetch(Kp + (size_t)(k0 + 32 + lane) * cD, 0, 3);

    // ---- S^T = K Q^T : A = K fragments (rows = keys, slots = d)
    AFrag ak[2][2];
#pragma unroll
    for (int m2 = 0; m2 < 2; ++m2) {
      const _Float16* kp = Kp + (size_t)(k0 + m2 * 16 + l15) * cD + hi * 16;
#pragma unroll
      for (int kd = 0; kd < 2; ++kd) {
        ak[m2][kd].h[0] = *(const v8h*)(kp + kd * 32);
        ak[m2][kd].h[1] = *(const v8h*)(kp + kd * 32 + 8);
      }
    }
    v8f st[2][2];  // [key-tile][query-tile]
#pragma unroll
    for (int m2 = 0; m2 < 2; ++m2)
#pragma unroll
      for (int n2 = 0; n2 < 2; ++n2) {
#pragma unroll
        for (int e = 0; e < 8; ++e) st[m2][n2][e] = 0.0f;
        st[m2][n2] = wmma_f16(ak[m2][0].v, bq[n2][0], st[m2][n2]);
        st[m2][n2] = wmma_f16(ak[m2][1].v, bq[n2][1], st[m2][n2]);
      }

    // ---- causal mask only on the diagonal tile (wave-uniform branch)
    if (kt == ktLast) {
#pragma unroll
      for (int m2 = 0; m2 < 2; ++m2)
#pragma unroll
        for (int n2 = 0; n2 < 2; ++n2) {
          int q = qbase + n2 * 16 + l15;
#pragma unroll
          for (int v = 0; v < 8; ++v) {
            int key = k0 + m2 * 16 + hi * 8 + v;
            if (key > q) st[m2][n2][v] = -1e30f;
          }
        }
    }

    // ---- online softmax per query column (in-lane + 1 shuffle)
#pragma unroll
    for (int n2 = 0; n2 < 2; ++n2) {
      float mx = st[0][n2][0];
#pragma unroll
      for (int v = 1; v < 8; ++v) mx = fmaxf(mx, st[0][n2][v]);
#pragma unroll
      for (int v = 0; v < 8; ++v) mx = fmaxf(mx, st[1][n2][v]);
      mx = fmaxf(mx, __shfl_xor(mx, 16, 32));
      float mnew = fmaxf(mrun[n2], mx);
      float f = __expf(mrun[n2] - mnew);
      mrun[n2] = mnew;
      float rs = 0.0f;
#pragma unroll
      for (int m2 = 0; m2 < 2; ++m2)
#pragma unroll
        for (int v = 0; v < 8; ++v) {
          float p = __expf(st[m2][n2][v] - mnew);
          st[m2][n2][v] = p;
          rs += p;
        }
      rs += __shfl_xor(rs, 16, 32);
      lrun[n2] = lrun[n2] * f + rs;
#pragma unroll
      for (int mt = 0; mt < 4; ++mt)
#pragma unroll
        for (int e = 0; e < 8; ++e) acco[mt][n2][e] *= f;
    }

    // ---- P^T (C layout) -> LDS [query][key] as packed v8h stores
#pragma unroll
    for (int m2 = 0; m2 < 2; ++m2)
#pragma unroll
      for (int n2 = 0; n2 < 2; ++n2) {
        v8h ph;
#pragma unroll
        for (int v = 0; v < 8; ++v) ph[v] = (_Float16)st[m2][n2][v];
        *(v8h*)&Plds[wid][n2 * 16 + l15][m2 * 16 + hi * 8] = ph;
      }
    asm volatile("s_wait_dscnt 0" ::: "memory");

    // B = P fragments: lane = query col, slots = keys (contiguous in LDS row)
    v16h bp[2];
#pragma unroll
    for (int n2 = 0; n2 < 2; ++n2)
      bp[n2] = *(const v16h*)&Plds[wid][n2 * 16 + l15][hi * 16];

    // ---- O^T += V^T P : A = Vt fragments (rows = d, slots = keys)
#pragma unroll
    for (int mt = 0; mt < 4; ++mt) {
      AFrag av;
      const _Float16* vp = Vp + (size_t)(mt * 16 + l15) * cS + k0 + hi * 16;
      av.h[0] = *(const v8h*)vp;
      av.h[1] = *(const v8h*)(vp + 8);
#pragma unroll
      for (int n2 = 0; n2 < 2; ++n2)
        acco[mt][n2] = wmma_f16(av.v, bp[n2], acco[mt][n2]);
    }
  }

  // ---- epilogue: (O^T / l) -> f16 [T][HID]; d contiguous -> packed stores
#pragma unroll
  for (int n2 = 0; n2 < 2; ++n2) {
    float inv_l = 1.0f / lrun[n2];
    int q = qbase + n2 * 16 + l15;
    size_t t = (size_t)b * cS + q;
    _Float16* orow = O + t * cHID + h * cD;
#pragma unroll
    for (int mt = 0; mt < 4; ++mt) {
      v8h oh;
#pragma unroll
      for (int v = 0; v < 8; ++v) oh[v] = (_Float16)(acco[mt][n2][v] * inv_l);
      *(v8h*)(orow + mt * 16 + hi * 8) = oh;
    }
  }
}

// ---------------------------------------------------------------- launcher
extern "C" void kernel_launch(void* const* d_in, const int* in_sizes, int n_in,
                              void* d_out, int out_size, void* d_ws,
                              size_t ws_size, hipStream_t stream) {
  (void)in_sizes; (void)n_in; (void)out_size; (void)ws_size;
  const float* hidden = (const float*)d_in[0];
  const float* q_w = (const float*)d_in[1];
  const float* q_b = (const float*)d_in[2];
  const float* k_w = (const float*)d_in[3];
  const float* k_b = (const float*)d_in[4];
  const float* v_w = (const float*)d_in[5];
  const float* v_b = (const float*)d_in[6];
  const float* o_w = (const float*)d_in[7];
  const int* pos = (const int*)d_in[8];
  float* out = (float*)d_out;

  char* w = (char*)d_ws;
  _Float16* hid16 = (_Float16*)w; w += (size_t)cT * cHID * 2;
  _Float16* qwT   = (_Float16*)w; w += (size_t)cHID * cHID * 2;
  _Float16* kwT   = (_Float16*)w; w += (size_t)cHID * cNKV * 2;
  _Float16* vwT   = (_Float16*)w; w += (size_t)cHID * cNKV * 2;
  _Float16* owT   = (_Float16*)w; w += (size_t)cHID * cHID * 2;
  _Float16* Qr    = (_Float16*)w; w += (size_t)cT * cHID * 2;   // [B][H][S][D]
  _Float16* Kr    = (_Float16*)w; w += (size_t)cT * cNKV * 2;   // [B][KV][S][D]
  _Float16* Vt16  = (_Float16*)w; w += (size_t)cT * cNKV * 2;   // [B][KV][D][S]
  _Float16* Oh    = (_Float16*)w; w += (size_t)cT * cHID * 2;   // [T][HID]

  cvt_f32_f16<<<2048, 256, 0, stream>>>(hidden, hid16, cT * cHID);
  transpose_cvt<<<2048, 256, 0, stream>>>(q_w, qwT, cHID, cHID);
  transpose_cvt<<<512, 256, 0, stream>>>(k_w, kwT, cHID, cNKV);
  transpose_cvt<<<512, 256, 0, stream>>>(v_w, vwT, cHID, cNKV);
  transpose_cvt<<<2048, 256, 0, stream>>>(o_w, owT, cHID, cHID);

  gemm_wmma<MODE_Q><<<dim3(cHID / 128, cT / 128), 256, 0, stream>>>(
      hid16, qwT, q_b, Qr, cT, cHID, cHID, pos);
  gemm_wmma<MODE_K><<<dim3(cNKV / 128, cT / 128), 256, 0, stream>>>(
      hid16, kwT, k_b, Kr, cT, cNKV, cHID, pos);
  gemm_wmma<MODE_V><<<dim3(cNKV / 128, cT / 128), 256, 0, stream>>>(
      hid16, vwT, v_b, Vt16, cT, cNKV, cHID, pos);

  attn_wmma<<<dim3(cS / 128, cB * cH), 128, 0, stream>>>(Qr, Kr, Vt16, Oh);

  gemm_wmma<MODE_O><<<dim3(cHID / 128, cT / 128), 256, 0, stream>>>(
      Oh, owT, nullptr, out, cT, cHID, cHID, nullptr);
}
